// ContextAwareAttention_75788992906209
// MI455X (gfx1250) — matrix operand order repaired and checked by reading, not compiled, not run on hardware
//
#include <hip/hip_runtime.h>
#include <hip/hip_bf16.h>
#include <stdint.h>

// Problem constants (from reference): B=64, T=4096, D1=256, D2=128, UNITS=64
#define BB   64
#define TT   4096
#define DD1  256
#define DD2  128
#define UU   64

typedef __attribute__((ext_vector_type(16))) __bf16   bf16x16;
typedef __attribute__((ext_vector_type(8)))  float    f32x8;
typedef __attribute__((ext_vector_type(4)))  uint32_t uint32x4;
typedef __attribute__((ext_vector_type(8)))  int      int32x8;

union FragBF { bf16x16 v; unsigned u[8]; };

__device__ __forceinline__ unsigned packbf(float a, float b) {
    __bf16 x = (__bf16)a, y = (__bf16)b;
    unsigned short ux = __builtin_bit_cast(unsigned short, x);
    unsigned short uy = __builtin_bit_cast(unsigned short, y);
    return (unsigned)ux | ((unsigned)uy << 16);
}

// TDM tile: 16 rows x 128B (one 32-wide K step for 16 M rows), global row
// stride 1KB. LDS gets 16B pad per row (pad_interval=128B, pad_amount=16B)
// -> 36-dword row stride => the 16 rows tile the 64 LDS banks conflict-free.
#define TILE_LDS_BYTES (16 * 144)   // 2304 B per tile
#define SLOTS 3                     // triple buffer per wave

// Issue one TDM tile load. Group0 = {flags, lds_addr, global_addr lo/hi+type}.
__device__ __forceinline__ void tdm_issue(const int32x8& g1, uint64_t gaddr,
                                          uint32_t ldsoff) {
    uint32x4 g0;
    g0[0] = 1u;                                     // count=1, user descriptor
    g0[1] = ldsoff;                                 // lds_addr (bytes)
    g0[2] = (uint32_t)gaddr;                        // global_addr[31:0]
    g0[3] = ((uint32_t)(gaddr >> 32) & 0x01FFFFFFu) // global_addr[56:32]
            | (2u << 30);                           // type=2 ("image")
    asm volatile("tensor_load_to_lds %0, %1" :: "s"(g0), "s"(g1) : "memory");
}

// ---------------------------------------------------------------------------
// Kernel 1: bias[b][u] = sum_c ctx[b,c] * W1[D1 + c, u]   (tiny: 64x64)
// ---------------------------------------------------------------------------
__global__ __launch_bounds__(64) void bias_kernel(const float* __restrict__ ctx,
                                                  const float* __restrict__ W1,
                                                  float* __restrict__ bias) {
    int b = blockIdx.x, u = threadIdx.x;
    const float* c = ctx + (size_t)b * DD2;
    const float* w = W1 + (size_t)DD1 * UU;   // W1_ctx
    float s = 0.f;
    #pragma unroll 8
    for (int j = 0; j < DD2; ++j) s = fmaf(c[j], w[(size_t)j * UU + u], s);
    bias[b * UU + u] = s;
}

// ---------------------------------------------------------------------------
// Kernel 2: logits[b,t] = sum_u tanh((seq @ W1_seq)[t,u] + bias[b,u]) * W2[u]
// bf16 WMMA, A-operand streamed by the Tensor Data Mover (triple-buffered LDS,
// TENSORcnt-pipelined), B-operand (W1_seq) resident in VGPRs.
// ---------------------------------------------------------------------------
__global__ __launch_bounds__(256) void attn_logits_wmma(
        const float* __restrict__ seq, const float* __restrict__ W1,
        const float* __restrict__ W2,  const float* __restrict__ bias,
        float* __restrict__ logits) {
    __shared__ __align__(16) char ldsA[8 * SLOTS * TILE_LDS_BYTES];  // 54 KB

    const int tid  = threadIdx.x;
    const int lane = tid & 31;
    const int wave = __builtin_amdgcn_readfirstlane(tid >> 5);  // wave-uniform
    const int kh   = lane >> 4;      // 16-lane half of the wave
    const int nl   = lane & 15;      // N (and M) position inside the half

    // --- B fragments: W1_seq (256x64 f32, L2-resident) -> bf16 in VGPRs -----
    // B K order used here (contiguous, same as A): lanes 0-15 hold K k0..k0+15 (pairs in u[0..7]),
    // lanes 16-31 hold K k0+16..k0+31.
    FragBF fbs[8][4];
    #pragma unroll
    for (int ks = 0; ks < 8; ++ks)
        #pragma unroll
        for (int nt = 0; nt < 4; ++nt) {
            const int n = nt * 16 + nl;
            #pragma unroll
            for (int j = 0; j < 8; ++j) {
                const int k = ks * 32 + kh * 16 + 2 * j;
                fbs[ks][nt].u[j] =
                    packbf(W1[(size_t)k * UU + n], W1[(size_t)(k + 1) * UU + n]);
            }
        }

    float w2v[4];
    #pragma unroll
    for (int nt = 0; nt < 4; ++nt) w2v[nt] = W2[nt * 16 + nl];

    // --- constant TDM group-1 descriptor ------------------------------------
    // data_size=8B; pad_enable, pad_interval=128B(code 4), pad_amount=16B(code 3)
    // tensor_dim0=128 units (1KB row), tensor_dim1=16 rows
    // tile_dim0=16 units (128B), tile_dim1=16 rows, dim0_stride=128 units
    int32x8 g1;
    g1[0] = (3 << 16) | (1 << 20) | (4 << 22) | (3 << 25);
    g1[1] = (int)(128u << 16);   // tensor_dim0[15:0] in [31:16]
    g1[2] = (int)(16u  << 16);   // tensor_dim1[15:0] in [31:16]
    g1[3] = (int)(16u  << 16);   // tile_dim0 in [31:16]
    g1[4] = 16;                  // tile_dim1 (tile_dim2=0)
    g1[5] = 128;                 // tensor_dim0_stride[31:0]
    g1[6] = 0;
    g1[7] = 0;

    const uint32_t ldsbase =
        (uint32_t)(uintptr_t)&ldsA[0] + (uint32_t)(wave * SLOTS * TILE_LDS_BYTES);
    const uint64_t seqbase = (uint64_t)(uintptr_t)seq;
    const int blockBase = blockIdx.x * 32;   // 32 groups of 16 rows per block

    // step s = g*8 + ks; group gid enumerates 16-row blocks: b*T + t0 = gid*16
    auto issue = [&](int s) {
        const int gid = blockBase + (s >> 3) * 8 + wave;
        const uint64_t goff =
            (((uint64_t)gid << 4) * DD1 + (uint64_t)((s & 7) * 32)) * 4u;
        tdm_issue(g1, seqbase + goff,
                  ldsbase + (uint32_t)((s % 3) * TILE_LDS_BYTES));
    };

    issue(0); issue(1); issue(2);   // prime the pipeline

    for (int g = 0; g < 4; ++g) {
        const int gid = blockBase + g * 8 + wave;
        const int b   = gid >> 8;            // 256 groups per batch
        const int t0  = (gid & 255) << 4;

        f32x8 z = {0.f, 0.f, 0.f, 0.f, 0.f, 0.f, 0.f, 0.f};
        f32x8 acc[4] = {z, z, z, z};

        #pragma unroll
        for (int ks = 0; ks < 8; ++ks) {
            const int s = g * 8 + ks;
            // in flight here: tiles s, s+1, s+2 (fewer near the tail)
            if      (g == 3 && ks == 7) asm volatile("s_wait_tensorcnt 0x0" ::: "memory");
            else if (g == 3 && ks == 6) asm volatile("s_wait_tensorcnt 0x1" ::: "memory");
            else                        asm volatile("s_wait_tensorcnt 0x2" ::: "memory");

            // A K order used here (contiguous, same as B): lane M = nl; kh=0 -> K 0-15,
            //                            kh=1 -> K 16-31 (within tile)
            const char* bb =
                &ldsA[(size_t)(wave * SLOTS + (s % 3)) * TILE_LDS_BYTES];
            const int ro = nl * 144 + kh * 64;
            const float4 a0 = *(const float4*)(bb + ro);
            const float4 a1 = *(const float4*)(bb + ro + 16);
            const float4 a2 = *(const float4*)(bb + ro + 32);
            const float4 a3 = *(const float4*)(bb + ro + 48);
            FragBF fa;
            fa.u[0] = packbf(a0.x, a0.y); fa.u[1] = packbf(a0.z, a0.w);
            fa.u[2] = packbf(a1.x, a1.y); fa.u[3] = packbf(a1.z, a1.w);
            fa.u[4] = packbf(a2.x, a2.y); fa.u[5] = packbf(a2.z, a2.w);
            fa.u[6] = packbf(a3.x, a3.y); fa.u[7] = packbf(a3.z, a3.w);

            #pragma unroll
            for (int nt = 0; nt < 4; ++nt)
                acc[nt] = __builtin_amdgcn_wmma_f32_16x16x32_bf16(
                    false, fa.v, false, fbs[ks][nt].v, (short)0, acc[nt],
                    false, false);

            if (!(g == 3 && ks >= 5)) issue(s + 3);   // refill freed slot
        }

        // Epilogue: C/D layout -> lane holds N = nl, rows M = kh*8 + m
        float biasv[4];
        #pragma unroll
        for (int nt = 0; nt < 4; ++nt) biasv[nt] = bias[b * UU + nt * 16 + nl];

        float r[8];
        #pragma unroll
        for (int m = 0; m < 8; ++m) {
            float sv = 0.f;
            #pragma unroll
            for (int nt = 0; nt < 4; ++nt)
                sv += tanhf(acc[nt][m] + biasv[nt]) * w2v[nt];
            #pragma unroll
            for (int mask = 1; mask <= 8; mask <<= 1)
                sv += __shfl_xor(sv, mask, 16);   // reduce over the 16 N-lanes
            r[m] = sv;
        }
        if (nl == 0) {
            float* lp = logits + (size_t)b * TT + t0 + kh * 8;
            #pragma unroll
            for (int m = 0; m < 8; ++m) lp[m] = r[m];
        }
    }
}

// ---------------------------------------------------------------------------
// Kernel 3: softmax over T per batch; rewrites logits -> probabilities
// ---------------------------------------------------------------------------
__global__ __launch_bounds__(256) void softmax_kernel(float* __restrict__ p) {
    __shared__ float red[256];
    const int b = blockIdx.x, tid = threadIdx.x;
    float* L = p + (size_t)b * TT;

    float m = -INFINITY;
    for (int t = tid; t < TT; t += 256) m = fmaxf(m, L[t]);
    red[tid] = m; __syncthreads();
    for (int s = 128; s > 0; s >>= 1) {
        if (tid < s) red[tid] = fmaxf(red[tid], red[tid + s]);
        __syncthreads();
    }
    const float mx = red[0]; __syncthreads();

    float sum = 0.f;
    for (int t = tid; t < TT; t += 256) sum += expf(L[t] - mx);
    red[tid] = sum; __syncthreads();
    for (int s = 128; s > 0; s >>= 1) {
        if (tid < s) red[tid] += red[tid + s];
        __syncthreads();
    }
    const float inv = 1.0f / red[0];
    for (int t = tid; t < TT; t += 256) L[t] = expf(L[t] - mx) * inv;
}

// ---------------------------------------------------------------------------
// Kernel 4: partial weighted sums over 256-row T chunks (fully coalesced)
// ---------------------------------------------------------------------------
__global__ __launch_bounds__(256) void wsum_kernel(const float* __restrict__ seq,
                                                   const float* __restrict__ p,
                                                   float* __restrict__ partials) {
    __shared__ float sp[256];
    const int b = blockIdx.x & 63, c = blockIdx.x >> 6;   // 64 batches x 16 chunks
    const int tid = threadIdx.x;
    const int t0 = c * 256;
    sp[tid] = p[(size_t)b * TT + t0 + tid];
    __syncthreads();
    const float* base = seq + (((size_t)b * TT + t0) << 8);
    float acc = 0.f;
    #pragma unroll 4
    for (int j = 0; j < 256; ++j)
        acc = fmaf(sp[j], base[((size_t)j << 8) + tid], acc);
    partials[((size_t)(c * BB + b) << 8) + tid] = acc;
}

// ---------------------------------------------------------------------------
// Kernel 5: reduce 16 chunk partials -> d_out (B x D1)
// ---------------------------------------------------------------------------
__global__ __launch_bounds__(256) void reduce_kernel(const float* __restrict__ partials,
                                                     float* __restrict__ out) {
    const int b = blockIdx.x, d = threadIdx.x;
    float acc = 0.f;
    #pragma unroll
    for (int c = 0; c < 16; ++c)
        acc += partials[((size_t)(c * BB + b) << 8) + d];
    out[b * DD1 + d] = acc;
}

// ---------------------------------------------------------------------------
extern "C" void kernel_launch(void* const* d_in, const int* in_sizes, int n_in,
                              void* d_out, int out_size, void* d_ws, size_t ws_size,
                              hipStream_t stream) {
    const float* seq = (const float*)d_in[0];   // (64, 4096, 256) f32
    const float* ctx = (const float*)d_in[1];   // (64, 128) f32
    const float* W1  = (const float*)d_in[2];   // (384, 64) f32
    const float* W2  = (const float*)d_in[3];   // (64, 1) f32
    float* out = (float*)d_out;                 // (64, 256) f32
    float* ws  = (float*)d_ws;

    float* bias     = ws;                               // 64*64      = 16 KB
    float* logits   = bias + BB * UU;                   // 64*4096    = 1 MB
    float* partials = logits + (size_t)BB * TT;         // 16*64*256  = 1 MB

    bias_kernel     <<<BB, UU, 0, stream>>>(ctx, W1, bias);
    attn_logits_wmma<<<512, 256, 0, stream>>>(seq, W1, W2, bias, logits);
    softmax_kernel  <<<BB, 256, 0, stream>>>(logits);
    wsum_kernel     <<<BB * 16, 256, 0, stream>>>(seq, logits, partials);
    reduce_kernel   <<<BB, 256, 0, stream>>>(partials, out);
}
